// SmallerNet_26620207301224
// MI455X (gfx1250) — compile-verified, matrix operand order checked by reading, not run on hardware
//
#include <hip/hip_runtime.h>

// ---------------------------------------------------------------------------
// CDNA5 (gfx1250): SAGEConv + MLP + pairwise L2
//   - scatter-mean via global f32 atomics (accumulator fully L2-resident)
//   - dense layers via v_wmma_f32_16x16x32_bf16, LDS-free:
//       A fragments loaded directly (per-lane row, two contiguous 8-float
//       chunks == exactly the WMMA 16-bit A layout), B pre-packed into
//       fragment order (one 32B vector load per lane per K-step)
//   - b128 streaming for the 400 MB [N,N] output
// ---------------------------------------------------------------------------

typedef __attribute__((ext_vector_type(16))) __bf16 v16bf;
typedef __attribute__((ext_vector_type(8)))  float  v8f;

// WMMA 16-bit operand layout (ISA 7.12.2):
//   lane L, slot pair v: K = ((v>>2)<<4) + ((L>=16)?8:0) + ((v&3)<<1)
// => per lane the 16 slots are K = {hi8..hi8+7, 16+hi8..16+hi8+7} ascending.
__device__ __forceinline__ int wmma_kbase(int v, int hi8) {
    return ((v >> 2) << 4) + hi8 + ((v & 3) << 1);
}

// Convert 4 float4 chunks (16 ascending K-values) into a bf16 A/B fragment.
__device__ __forceinline__ v16bf frag_from_f32(float4 f0, float4 f1,
                                               float4 f2, float4 f3, float s) {
    v16bf a;
    a[0]  = (__bf16)(f0.x * s); a[1]  = (__bf16)(f0.y * s);
    a[2]  = (__bf16)(f0.z * s); a[3]  = (__bf16)(f0.w * s);
    a[4]  = (__bf16)(f1.x * s); a[5]  = (__bf16)(f1.y * s);
    a[6]  = (__bf16)(f1.z * s); a[7]  = (__bf16)(f1.w * s);
    a[8]  = (__bf16)(f2.x * s); a[9]  = (__bf16)(f2.y * s);
    a[10] = (__bf16)(f2.z * s); a[11] = (__bf16)(f2.w * s);
    a[12] = (__bf16)(f3.x * s); a[13] = (__bf16)(f3.y * s);
    a[14] = (__bf16)(f3.z * s); a[15] = (__bf16)(f3.w * s);
    return a;
}

// ------------------------------ utility kernels ----------------------------

__global__ void zero_f32(float* __restrict__ p, long n) {
    long i = (long)blockIdx.x * blockDim.x + threadIdx.x;
    if (i < n) p[i] = 0.0f;
}

// Repack a row-major fp32 weight W[K,Nd] into bf16 WMMA B-fragment order:
// Bp laid out as [nt][kt][lane(32)][slot(16)] so a wave loads one K-tile of
// B with a single v16bf (32B) load per lane.
__global__ void repack_w_bf16(const float* __restrict__ W, __bf16* __restrict__ Bp,
                              int K, int Nd) {
    int idx = blockIdx.x * blockDim.x + threadIdx.x;
    if (idx >= K * Nd) return;
    int tile   = idx >> 9;          // 512 elements per 32x16 tile
    int within = idx & 511;
    int lane   = within >> 4;
    int s      = within & 15;
    int ktiles = K >> 5;
    int nt     = tile / ktiles;
    int kt     = tile - nt * ktiles;
    int n      = (nt << 4) + (lane & 15);
    int hi8    = (lane >> 4) << 3;
    int k      = (kt << 5) + wmma_kbase(s >> 1, hi8) + (s & 1);
    Bp[idx] = (__bf16)W[(size_t)k * Nd + n];
}

// ------------------------- scatter-mean aggregation ------------------------

__global__ void aggr_kernel(const float* __restrict__ x,
                            const long long* __restrict__ ei,
                            float* __restrict__ sums, float* __restrict__ cnt,
                            int E, int DIN) {
    int e = blockIdx.x;
    int src = (int)ei[e];
    int dst = (int)ei[(long)E + e];
    const float* xr = x + (size_t)src * DIN;
    float* sr = sums + (size_t)dst * DIN;
    for (int d = threadIdx.x; d < DIN; d += blockDim.x)
        unsafeAtomicAdd(&sr[d], xr[d]);
    if (threadIdx.x == 0) unsafeAtomicAdd(&cnt[dst], 1.0f);
}

// ------------------------- generic bf16 WMMA GEMM --------------------------
// C[M,Nd] = act( A[M,K] @ W + bias ). 8 waves/block, one 16x16 tile/wave.
// LDS-free: lane L reads its A row directly; lanes m and m+16 jointly cover
// each 128B row chunk, so the wave's loads are fully cacheline-covering.

__global__ void gemm_wmma_bf16(const float* __restrict__ A,
                               const __bf16* __restrict__ Bp,
                               const float* __restrict__ bias,
                               float* __restrict__ C,
                               int M, int Nd, int K, int do_relu) {
    const int lane  = threadIdx.x & 31;
    const int wv    = threadIdx.x >> 5;
    const int mtile = blockIdx.y * 8 + wv;
    if (mtile >= (M >> 4)) return;
    const int m0     = mtile << 4;
    const int n0     = blockIdx.x << 4;
    const int ln15   = lane & 15;
    const int hi8    = (lane >> 4) << 3;
    const int ktiles = K >> 5;
    const v16bf* Bt  = (const v16bf*)Bp + (size_t)blockIdx.x * ktiles * 32;
    const float* arow = A + (size_t)(m0 + ln15) * K + hi8;

    v8f acc = {};
    for (int kt = 0; kt < ktiles; ++kt) {
        const float4* c = (const float4*)(arow + (kt << 5));
        __builtin_prefetch(arow + ((kt + 1) << 5), 0, 3);   // speculative
        float4 f0 = c[0], f1 = c[1], f2 = c[4], f3 = c[5];
        v16bf a = frag_from_f32(f0, f1, f2, f3, 1.0f);
        v16bf b = Bt[kt * 32 + lane];
        acc = __builtin_amdgcn_wmma_f32_16x16x32_bf16(false, a, false, b,
                                                      (short)0, acc, false, false);
    }

    const float bb    = bias[n0 + ln15];
    const int   rbase = (lane >> 4) << 3;
    #pragma unroll
    for (int r = 0; r < 8; ++r) {
        float v = acc[r] + bb;
        if (do_relu) v = fmaxf(v, 0.0f);
        C[(size_t)(m0 + rbase + r) * Nd + n0 + ln15] = v;
    }
}

// --------------------------- fused SAGE layer ------------------------------
// h1 = relu( x @ Wr + (sums/max(cnt,1)) @ Wl + bl ), K=512, Nd=256.
// Two GEMM phases share one WMMA accumulator; the mean scale is a per-lane
// constant (each lane owns one row), applied during bf16 conversion.

__global__ void sage_wmma(const float* __restrict__ x,
                          const float* __restrict__ sums,
                          const float* __restrict__ cnt,
                          const __bf16* __restrict__ Wr_p,
                          const __bf16* __restrict__ Wl_p,
                          const float* __restrict__ bl,
                          float* __restrict__ h1, int M) {
    const int K = 512, Nd = 256, ktiles = 16;
    const int lane  = threadIdx.x & 31;
    const int wv    = threadIdx.x >> 5;
    const int mtile = blockIdx.y * 8 + wv;
    if (mtile >= (M >> 4)) return;
    const int m0   = mtile << 4;
    const int n0   = blockIdx.x << 4;
    const int ln15 = lane & 15;
    const int hi8  = (lane >> 4) << 3;
    const float rscale = 1.0f / fmaxf(cnt[m0 + ln15], 1.0f);

    v8f acc = {};
    #pragma unroll 1
    for (int ph = 0; ph < 2; ++ph) {
        const float* arow = (ph ? sums : x) + (size_t)(m0 + ln15) * K + hi8;
        const float  sc   = ph ? rscale : 1.0f;
        const v16bf* Bt   = (const v16bf*)(ph ? Wl_p : Wr_p)
                            + (size_t)blockIdx.x * ktiles * 32;
        for (int kt = 0; kt < ktiles; ++kt) {
            const float4* c = (const float4*)(arow + (kt << 5));
            __builtin_prefetch(arow + ((kt + 1) << 5), 0, 3);
            float4 f0 = c[0], f1 = c[1], f2 = c[4], f3 = c[5];
            v16bf a = frag_from_f32(f0, f1, f2, f3, sc);
            v16bf b = Bt[kt * 32 + lane];
            acc = __builtin_amdgcn_wmma_f32_16x16x32_bf16(false, a, false, b,
                                                          (short)0, acc, false, false);
        }
    }

    const float bb    = bl[n0 + ln15];
    const int   rbase = (lane >> 4) << 3;
    #pragma unroll
    for (int r = 0; r < 8; ++r) {
        float v = fmaxf(acc[r] + bb, 0.0f);
        h1[(size_t)(m0 + rbase + r) * Nd + n0 + ln15] = v;
    }
}

// --------------------------- 32 -> 3 head + norms --------------------------

__global__ void head_kernel(const float* __restrict__ h4,
                            const float* __restrict__ W3,
                            const float* __restrict__ b3,
                            float* __restrict__ p, int M) {
    int i = blockIdx.x * blockDim.x + threadIdx.x;
    if (i >= M) return;
    float o0 = b3[0], o1 = b3[1], o2 = b3[2];
    const float* hr = h4 + (size_t)i * 32;
    #pragma unroll
    for (int k = 0; k < 32; ++k) {
        float v = hr[k];
        o0 += v * W3[k * 3 + 0];
        o1 += v * W3[k * 3 + 1];
        o2 += v * W3[k * 3 + 2];
    }
    float4 r;
    r.x = o0; r.y = o1; r.z = o2;
    r.w = o0 * o0 + o1 * o1 + o2 * o2;
    ((float4*)p)[i] = r;
}

// ------------------------- pairwise L2 (N x N out) -------------------------
// Block covers 1024 j-columns for one i-row: stage the 1024 {h, |h|^2} points
// in LDS, then each thread emits a 16B (float4) coalesced store.

__global__ void pairwise_kernel(const float4* __restrict__ p4,
                                float* __restrict__ out, int M) {
    __shared__ float4 Pj[1024];
    const int tid  = threadIdx.x;
    const int base = blockIdx.x << 10;
    for (int t = tid; t < 1024; t += 256) {
        int j = base + t;
        Pj[t] = (j < M) ? p4[j] : make_float4(0.f, 0.f, 0.f, 0.f);
    }
    __syncthreads();

    const int i = blockIdx.y;
    const float4 pi = p4[i];
    const int j0 = base + (tid << 2);
    if (j0 >= M) return;

    float o[4];
    #pragma unroll
    for (int k = 0; k < 4; ++k) {
        float4 pj = Pj[(tid << 2) + k];
        float d2 = pi.w + pj.w - 2.0f * (pi.x * pj.x + pi.y * pj.y + pi.z * pj.z);
        d2 = fmaxf(d2, 0.0f);
        o[k] = (d2 > 0.0f) ? sqrtf(d2) : 0.0f;
    }
    size_t off = (size_t)i * M + j0;
    if (j0 + 3 < M) {
        *(float4*)&out[off] = make_float4(o[0], o[1], o[2], o[3]);
    } else {
        for (int k = 0; k < 4 && j0 + k < M; ++k) out[off + k] = o[k];
    }
}

// ------------------------------- launcher ----------------------------------

extern "C" void kernel_launch(void* const* d_in, const int* in_sizes, int n_in,
                              void* d_out, int out_size, void* d_ws, size_t ws_size,
                              hipStream_t stream) {
    const float*     x   = (const float*)d_in[0];
    const long long* ei  = (const long long*)d_in[1];   // int64 edge_index [2,E]
    const float*     W_l = (const float*)d_in[2];
    const float*     b_l = (const float*)d_in[3];
    const float*     W_r = (const float*)d_in[4];
    const float*     Wa  = (const float*)d_in[5];
    const float*     ba  = (const float*)d_in[6];
    const float*     W1  = (const float*)d_in[7];
    const float*     b1  = (const float*)d_in[8];
    const float*     W2  = (const float*)d_in[9];
    const float*     b2  = (const float*)d_in[10];
    const float*     W3  = (const float*)d_in[11];
    const float*     b3  = (const float*)d_in[12];

    const int DIN = 512;
    const int Nn  = in_sizes[0] / DIN;    // 10000
    const int E   = in_sizes[1] / 2;      // 160000
    float* out = (float*)d_out;

    // workspace bump allocator (256B aligned)
    char*  w   = (char*)d_ws;
    size_t off = 0;
    auto alloc = [&](size_t bytes) -> void* {
        void* ptr = w + off;
        off = (off + bytes + 255) & ~(size_t)255;
        return ptr;
    };
    float*  sums = (float*)alloc((size_t)Nn * DIN * 4);   // 20.48 MB
    float*  cnt  = (float*)alloc((size_t)Nn * 4);         // contiguous after sums
    float*  h1   = (float*)alloc((size_t)Nn * 256 * 4);
    float*  h2   = (float*)alloc((size_t)Nn * 128 * 4);
    float*  h3   = (float*)alloc((size_t)Nn * 64 * 4);
    float*  h4   = (float*)alloc((size_t)Nn * 32 * 4);
    float*  pvec = (float*)alloc((size_t)Nn * 4 * 4);
    __bf16* Wl_p = (__bf16*)alloc((size_t)512 * 256 * 2);
    __bf16* Wr_p = (__bf16*)alloc((size_t)512 * 256 * 2);
    __bf16* Wa_p = (__bf16*)alloc((size_t)256 * 128 * 2);
    __bf16* W1_p = (__bf16*)alloc((size_t)128 * 64 * 2);
    __bf16* W2_p = (__bf16*)alloc((size_t)64 * 32 * 2);
    (void)ws_size; (void)n_in; (void)out_size;

    // 1) zero accumulators (sums + cnt are contiguous)
    {
        long nz = (long)Nn * DIN + Nn;
        zero_f32<<<dim3((unsigned)((nz + 255) / 256)), 256, 0, stream>>>(sums, nz);
    }

    // 2) repack weights fp32 -> bf16 WMMA fragment order (tiny, once/launch)
    repack_w_bf16<<<(512 * 256 + 255) / 256, 256, 0, stream>>>(W_l, Wl_p, 512, 256);
    repack_w_bf16<<<(512 * 256 + 255) / 256, 256, 0, stream>>>(W_r, Wr_p, 512, 256);
    repack_w_bf16<<<(256 * 128 + 255) / 256, 256, 0, stream>>>(Wa, Wa_p, 256, 128);
    repack_w_bf16<<<(128 * 64 + 255) / 256, 256, 0, stream>>>(W1, W1_p, 128, 64);
    repack_w_bf16<<<(64 * 32 + 255) / 256, 256, 0, stream>>>(W2, W2_p, 64, 32);

    // 3) scatter-mean numerators/denominators (L2-resident atomics)
    aggr_kernel<<<E, 256, 0, stream>>>(x, ei, sums, cnt, E, DIN);

    // 4) dense chain via WMMA (8 M-tiles per block)
    const int gy = (Nn / 16 + 7) / 8;
    sage_wmma<<<dim3(256 / 16, gy), 256, 0, stream>>>(x, sums, cnt, Wr_p, Wl_p, b_l, h1, Nn);
    gemm_wmma_bf16<<<dim3(128 / 16, gy), 256, 0, stream>>>(h1, Wa_p, ba, h2, Nn, 128, 256, 1);
    gemm_wmma_bf16<<<dim3(64 / 16, gy), 256, 0, stream>>>(h2, W1_p, b1, h3, Nn, 64, 128, 1);
    gemm_wmma_bf16<<<dim3(32 / 16, gy), 256, 0, stream>>>(h3, W2_p, b2, h4, Nn, 32, 64, 1);

    // 5) 32->3 head + squared norms
    head_kernel<<<(Nn + 255) / 256, 256, 0, stream>>>(h4, W3, b3, pvec, Nn);

    // 6) pairwise L2 -> [N, N]: LDS-staged, b128 stream-out (400 MB)
    pairwise_kernel<<<dim3((Nn + 1023) / 1024, Nn), 256, 0, stream>>>(
        (const float4*)pvec, out, Nn);
}
